// MGGNN_53747220742754
// MI455X (gfx1250) — matrix operand order repaired and checked by reading, not compiled, not run on hardware
//
#include <hip/hip_runtime.h>

// ---------------------------------------------------------------------------
// MI455X (gfx1250) implementation of the multigrid-GNN forward pass.
// Dense layers: v_wmma_f32_16x16x32_bf16 (bf16 A/B, f32 accum).
// Weights pre-packed into WMMA B-fragment order (32B contiguous per lane).
// Multi-layer MLPs fused per 16-row wave tile with LDS staging.
// segment_sum -> global_atomic_add_f32.
//
// Assumed d_in layout (setup_inputs() dict insertion order, params flattened
// leaf-by-leaf in insertion order):
//   [0] x0 [N0,1]  [1] edge_attr0 [E0,2]  [2] edge_attr1 [E1,1]  [3] r_vals [EF]
//   [4..81] 78 param leaves
//   [82] edge_index0 [2,E0] [83] edge_index1 [2,E1]
//   [84] f2c_fine [EF] [85] f2c_coarse [EF] [86] mask_row [EM] [87] mask_col [EM]
// d_out = concat(out [EM], out_R [EF]) fp32.
// ---------------------------------------------------------------------------

#define LDP 68  // LDS row pitch (floats); 68*4 = 272 B = 16-byte aligned

typedef __bf16 bhalf;
typedef __attribute__((ext_vector_type(16))) __bf16 v16bf;
typedef __attribute__((ext_vector_type(8)))  float  v8f;

static __device__ __forceinline__ v8f wmma_bf16(v16bf a, v16bf b, v8f c) {
  // (neg_a, A, neg_b, B, c_mod, C, reuse_a, reuse_b)
  return __builtin_amdgcn_wmma_f32_16x16x32_bf16(false, a, false, b, (short)0, c,
                                                 false, false);
}

static __device__ __forceinline__ v8f vzero8() {
  v8f a;
#pragma unroll
  for (int i = 0; i < 8; ++i) a[i] = 0.f;
  return a;
}

// A fragment (16x32 bf16): lane = row (lane&15), lane-half selects K subgroups.
// element e -> K = k0 + (e&7) + 8*half + (e>=8 ? 16 : 0). 16B-aligned float4 loads.
static __device__ __forceinline__ v16bf load_a_f32(const float* row, int k0, int half) {
  const float4 f0 = *(const float4*)(row + k0 + half * 8);
  const float4 f1 = *(const float4*)(row + k0 + half * 8 + 4);
  const float4 f2 = *(const float4*)(row + k0 + 16 + half * 8);
  const float4 f3 = *(const float4*)(row + k0 + 16 + half * 8 + 4);
  v16bf a;
  a[0] = (__bf16)f0.x;  a[1] = (__bf16)f0.y;  a[2]  = (__bf16)f0.z;  a[3]  = (__bf16)f0.w;
  a[4] = (__bf16)f1.x;  a[5] = (__bf16)f1.y;  a[6]  = (__bf16)f1.z;  a[7]  = (__bf16)f1.w;
  a[8] = (__bf16)f2.x;  a[9] = (__bf16)f2.y;  a[10] = (__bf16)f2.z;  a[11] = (__bf16)f2.w;
  a[12] = (__bf16)f3.x; a[13] = (__bf16)f3.y; a[14] = (__bf16)f3.z;  a[15] = (__bf16)f3.w;
  return a;
}

// Pre-packed B fragment: one contiguous 32-byte blob per lane per (ctIdx, kc).
static __device__ __forceinline__ v16bf load_bp(const bhalf* Wp, int nK, int ctIdx,
                                                int kc, int lane) {
  return *(const v16bf*)(Wp + (((size_t)ctIdx * nK + kc) * 32 + lane) * 16);
}

// 64->64 dense layer, LDS tile -> LDS tile (wave-private), bias + optional relu.
// Wp packed for K=64 (nK=2), N=64.
static __device__ __forceinline__ void gemm64_lds(const float* smIn, float* smOut,
                                                  const bhalf* Wp, const float* bias,
                                                  int cl, int half, int lane, bool relu) {
  const float* xr = smIn + cl * LDP;
  for (int ct = 0; ct < 4; ++ct) {
    v8f acc = vzero8();
    for (int kc = 0; kc < 2; ++kc)
      acc = wmma_bf16(load_a_f32(xr, kc * 32, half), load_bp(Wp, 2, ct, kc, lane), acc);
    int col = ct * 16 + cl;
    float bb = bias[col];
#pragma unroll
    for (int v = 0; v < 8; ++v) {
      float val = acc[v] + bb;
      if (relu) val = fmaxf(val, 0.f);
      smOut[(v + 8 * half) * LDP + col] = val;
    }
  }
}

// ----------------------------- utility kernels -----------------------------

__global__ void k_zero(float* p, long long n) {
  long long i = (long long)blockIdx.x * blockDim.x + threadIdx.x;
  long long st = (long long)gridDim.x * blockDim.x;
  for (; i < n; i += st) p[i] = 0.f;
}

__global__ void k_relu(float* p, long long n) {
  long long i = (long long)blockIdx.x * blockDim.x + threadIdx.x;
  long long st = (long long)gridDim.x * blockDim.x;
  for (; i < n; i += st) p[i] = fmaxf(p[i], 0.f);
}

// Pack W[K,N] (f32 row-major) into WMMA B-fragment order (bf16):
// dst[(((ct*nK)+kc)*32 + lane)*16 + e] = bf16(W[(kc*32 + 8*(lane>>4) + (e&7) + 16*(e>>3))*N + ct*16 + (lane&15)])
__global__ void k_packB(const float* W, bhalf* dst, int K, int N) {
  int i = blockIdx.x * blockDim.x + threadIdx.x;
  if (i >= K * N) return;
  int nK = K / 32;
  int e = i & 15;
  int lane = (i >> 4) & 31;
  int blk = i >> 9;
  int kc = blk % nK;
  int ct = blk / nK;
  int k = kc * 32 + 8 * (lane >> 4) + (e & 7) + ((e >> 3) << 4);
  int col = ct * 16 + (lane & 15);
  dst[i] = (__bf16)W[(size_t)k * N + col];
}

// x1[c] += x0[f] * r  (coarse node features = R^T x)
__global__ void k_scat_x1(const float* x0, const float* r, const int* fi, const int* ci,
                          float* x1, int E) {
  int e = blockIdx.x * blockDim.x + threadIdx.x;
  if (e < E) atomicAdd(&x1[ci[e]], x0[fi[e]] * r[e]);
}

// c[n,128] = [a[n,64] | b[n,64]]
__global__ void k_concat(const float* a, const float* b, float* c, long long n) {
  long long i = (long long)blockIdx.x * blockDim.x + threadIdx.x;
  if (i >= n * 128) return;
  long long rr = i >> 7;
  int ch = (int)(i & 127);
  c[i] = (ch < 64) ? a[rr * 64 + ch] : b[rr * 64 + (ch - 64)];
}

// Y[dst] += X[src] * w   (128 channels, one thread per (edge, channel))
__global__ void k_spmv(const float* X, const int* src, const int* dst, const float* w,
                       float* Y, long long E) {
  long long i = (long long)blockIdx.x * blockDim.x + threadIdx.x;
  if (i >= E * 128) return;
  long long e = i >> 7;
  int c = (int)(i & 127);
  atomicAdd(&Y[(size_t)dst[e] * 128 + c], X[(size_t)src[e] * 128 + c] * w[e]);
}

__global__ void k_colsum(const float* v, const int* idx, float* acc, int E) {
  int e = blockIdx.x * blockDim.x + threadIdx.x;
  if (e < E) atomicAdd(&acc[idx[e]], v[e]);
}

__global__ void k_rnorm(const float* v, const int* idx, const float* acc, float* out, int E) {
  int e = blockIdx.x * blockDim.x + threadIdx.x;
  if (e < E) out[e] = v[e] / acc[idx[e]];
}

// InstanceNorm(affine=False), whole graph = one instance, per channel
__global__ void k_inorm_stats(const float* X, int n, float* stats) {
  int t = blockIdx.x * blockDim.x + threadIdx.x;
  int c = t & 63;
  int r0 = t >> 6;
  int stride = (gridDim.x * blockDim.x) >> 6;
  float s = 0.f, sq = 0.f;
  for (int r = r0; r < n; r += stride) {
    float v = X[(size_t)r * 64 + c];
    s += v; sq += v * v;
  }
  atomicAdd(&stats[c], s);
  atomicAdd(&stats[64 + c], sq);
}

__global__ void k_inorm_apply(float* X, int n, const float* stats) {
  long long i = (long long)blockIdx.x * blockDim.x + threadIdx.x;
  if (i >= (long long)n * 64) return;
  int c = (int)(i & 63);
  float mu = stats[c] / (float)n;
  float var = stats[64 + c] / (float)n - mu * mu;
  X[i] = (X[i] - mu) * rsqrtf(var + 1e-5f);
}

// ----------------------------- WMMA kernels --------------------------------

// Plain GEMM: Y[M,N] = X[M,K] @ W + b  (K in {64,128}, N multiple of 16)
__global__ void k_gemm(const float* X, const bhalf* Wp, const float* b, float* Y,
                       int M, int K, int N, int relu) {
  int wave = (blockIdx.x * blockDim.x + threadIdx.x) >> 5;
  int lane = threadIdx.x & 31;
  int row0 = wave * 16;
  if (row0 >= M) return;
  int half = lane >> 4, cl = lane & 15;
  int nK = K / 32;
  bool full = (row0 + 16 <= M);
  int arow = row0 + cl; if (arow >= M) arow = M - 1;
  const float* xr = X + (size_t)arow * K;
  for (int ct = 0; ct < N / 16; ++ct) {
    v8f acc = vzero8();
    for (int kc = 0; kc < nK; ++kc)
      acc = wmma_bf16(load_a_f32(xr, kc * 32, half), load_bp(Wp, nK, ct, kc, lane), acc);
    int col = ct * 16 + cl;
    float bb = b[col];
    if (full) {
#pragma unroll
      for (int v = 0; v < 8; ++v) {
        float val = acc[v] + bb;
        if (relu) val = fmaxf(val, 0.f);
        Y[(size_t)(row0 + v + 8 * half) * N + col] = val;
      }
    } else {
#pragma unroll
      for (int v = 0; v < 8; ++v) {
        int r = row0 + v + 8 * half;
        if (r < M) {
          float val = acc[v] + bb;
          if (relu) val = fmaxf(val, 0.f);
          Y[(size_t)r * N + col] = val;
        }
      }
    }
  }
}

// TAGConv combine: Y = relu(C@W0 + T1@W1 + T2@W2 + b0+b1+b2), K=128, N=64
__global__ void k_tag_out(const float* C, const float* T1, const float* T2,
                          const bhalf* W0, const bhalf* W1, const bhalf* W2,
                          const float* b0, const float* b1, const float* b2,
                          float* Y, int M) {
  int wave = (blockIdx.x * blockDim.x + threadIdx.x) >> 5;
  int lane = threadIdx.x & 31;
  int row0 = wave * 16;
  if (row0 >= M) return;
  int half = lane >> 4, cl = lane & 15;
  bool full = (row0 + 16 <= M);
  int arow = row0 + cl; if (arow >= M) arow = M - 1;
  const float* r0p = C  + (size_t)arow * 128;
  const float* r1p = T1 + (size_t)arow * 128;
  const float* r2p = T2 + (size_t)arow * 128;
  for (int ct = 0; ct < 4; ++ct) {
    v8f acc = vzero8();
    for (int kc = 0; kc < 4; ++kc)
      acc = wmma_bf16(load_a_f32(r0p, kc * 32, half), load_bp(W0, 4, ct, kc, lane), acc);
    for (int kc = 0; kc < 4; ++kc)
      acc = wmma_bf16(load_a_f32(r1p, kc * 32, half), load_bp(W1, 4, ct, kc, lane), acc);
    for (int kc = 0; kc < 4; ++kc)
      acc = wmma_bf16(load_a_f32(r2p, kc * 32, half), load_bp(W2, 4, ct, kc, lane), acc);
    int col = ct * 16 + cl;
    float bb = b0[col] + b1[col] + b2[col];
    if (full) {
#pragma unroll
      for (int v = 0; v < 8; ++v)
        Y[(size_t)(row0 + v + 8 * half) * 64 + col] = fmaxf(acc[v] + bb, 0.f);
    } else {
#pragma unroll
      for (int v = 0; v < 8; ++v) {
        int r = row0 + v + 8 * half;
        if (r < M) Y[(size_t)r * 64 + col] = fmaxf(acc[v] + bb, 0.f);
      }
    }
  }
}

// pre_node: 1->64 relu -> 64->64 relu (WMMA) -> 64->64 linear (WMMA)
__global__ void k_pre_node(const float* x, const float* W0, const float* b0,
                           const bhalf* W1, const float* b1,
                           const bhalf* W2, const float* b2, float* H, int M) {
  __shared__ float smem[4 * 2 * 16 * LDP];
  int wv = threadIdx.x >> 5, lane = threadIdx.x & 31;
  int row0 = (blockIdx.x * 4 + wv) * 16;
  int half = lane >> 4, cl = lane & 15;
  bool full = (row0 + 16 <= M);
  float* smA = smem + (wv * 2) * (16 * LDP);
  float* smB = smA + 16 * LDP;
  for (int t = lane; t < 16 * 64; t += 32) {
    int r = t >> 6, j = t & 63;
    int e = row0 + r; if (e >= M) e = M - 1;
    smA[r * LDP + j] = fmaxf(x[e] * W0[j] + b0[j], 0.f);
  }
  __syncthreads();
  gemm64_lds(smA, smB, W1, b1, cl, half, lane, true);
  __syncthreads();
  const float* xr = smB + cl * LDP;
  for (int ct = 0; ct < 4; ++ct) {
    v8f acc = vzero8();
    for (int kc = 0; kc < 2; ++kc)
      acc = wmma_bf16(load_a_f32(xr, kc * 32, half), load_bp(W2, 2, ct, kc, lane), acc);
    int col = ct * 16 + cl;
    float bb = b2[col];
    if (full) {
#pragma unroll
      for (int v = 0; v < 8; ++v)
        H[(size_t)(row0 + v + 8 * half) * 64 + col] = acc[v] + bb;
    } else {
#pragma unroll
      for (int v = 0; v < 8; ++v) {
        int r = row0 + v + 8 * half;
        if (r < M) H[(size_t)r * 64 + col] = acc[v] + bb;
      }
    }
  }
}

// fused edge-weight chain for fine edges:
// attr[E,2] -> 2:64 relu -> 64:64 relu(WMMA) -> 64:1 -> 1:64 relu -> 64:64 relu(WMMA) -> 64:1
__global__ void k_w0_fused(const float* attr,
                           const float* W0a, const float* b0a, const bhalf* W1a,
                           const float* b1a, const float* W2a, const float* b2a,
                           const float* W0b, const float* b0b, const bhalf* W1b,
                           const float* b1b, const float* W2b, const float* b2b,
                           float* w, int E) {
  __shared__ float smem[4 * 2 * 16 * LDP];
  __shared__ float eaRow[4][16];
  int wv = threadIdx.x >> 5, lane = threadIdx.x & 31;
  int row0 = (blockIdx.x * 4 + wv) * 16;
  int half = lane >> 4, cl = lane & 15;
  float* smA = smem + (wv * 2) * (16 * LDP);
  float* smB = smA + 16 * LDP;
  for (int t = lane; t < 16 * 64; t += 32) {
    int r = t >> 6, j = t & 63;
    int e = row0 + r; if (e >= E) e = E - 1;
    float a0 = attr[(size_t)e * 2 + 0], a1 = attr[(size_t)e * 2 + 1];
    smA[r * LDP + j] = fmaxf(a0 * W0a[j] + a1 * W0a[64 + j] + b0a[j], 0.f);
  }
  __syncthreads();
  gemm64_lds(smA, smB, W1a, b1a, cl, half, lane, true);
  __syncthreads();
  if (lane < 16) {
    float s = b2a[0];
    for (int j = 0; j < 64; ++j) s += smB[lane * LDP + j] * W2a[j];
    eaRow[wv][lane] = s;
  }
  __syncthreads();
  for (int t = lane; t < 16 * 64; t += 32) {
    int r = t >> 6, j = t & 63;
    smA[r * LDP + j] = fmaxf(eaRow[wv][r] * W0b[j] + b0b[j], 0.f);
  }
  __syncthreads();
  gemm64_lds(smA, smB, W1b, b1b, cl, half, lane, true);
  __syncthreads();
  if (lane < 16) {
    int e = row0 + lane;
    if (e < E) {
      float s = b2b[0];
      for (int j = 0; j < 64; ++j) s += smB[lane * LDP + j] * W2b[j];
      w[e] = s;
    }
  }
}

// pre_edge only: x[E,1] -> 1:64 relu -> 64:64 relu(WMMA) -> 64:1
__global__ void k_pe_fused(const float* x, const float* W0, const float* b0,
                           const bhalf* W1, const float* b1,
                           const float* W2, const float* b2, float* w, int E) {
  __shared__ float smem[4 * 2 * 16 * LDP];
  int wv = threadIdx.x >> 5, lane = threadIdx.x & 31;
  int row0 = (blockIdx.x * 4 + wv) * 16;
  int half = lane >> 4, cl = lane & 15;
  float* smA = smem + (wv * 2) * (16 * LDP);
  float* smB = smA + 16 * LDP;
  for (int t = lane; t < 16 * 64; t += 32) {
    int r = t >> 6, j = t & 63;
    int e = row0 + r; if (e >= E) e = E - 1;
    smA[r * LDP + j] = fmaxf(x[e] * W0[j] + b0[j], 0.f);
  }
  __syncthreads();
  gemm64_lds(smA, smB, W1, b1, cl, half, lane, true);
  __syncthreads();
  if (lane < 16) {
    int e = row0 + lane;
    if (e < E) {
      float s = b2[0];
      for (int j = 0; j < 64; ++j) s += smB[lane * LDP + j] * W2[j];
      w[e] = s;
    }
  }
}

// PlainMP message+aggregate: per edge cat(Xd[di], Xs[si]) -> 128:64 relu -> 64:64
// -> atomicAdd into Acc[di]
__global__ void k_plainmp(const float* Xd, const int* di, const float* Xs, const int* si,
                          const bhalf* W0, const float* b0,
                          const bhalf* W1, const float* b1, float* Acc, int E) {
  __shared__ float smem[4 * 16 * LDP];
  int wv = threadIdx.x >> 5, lane = threadIdx.x & 31;
  int row0 = (blockIdx.x * 4 + wv) * 16;
  int half = lane >> 4, cl = lane & 15;
  bool full = (row0 + 16 <= E);
  float* smA = smem + wv * (16 * LDP);
  int e = row0 + cl; if (e >= E) e = E - 1;
  const float* pd = Xd + (size_t)di[e] * 64;
  const float* ps = Xs + (size_t)si[e] * 64;
  for (int ct = 0; ct < 4; ++ct) {
    v8f acc = vzero8();
    int col = ct * 16 + cl;
    for (int kc = 0; kc < 4; ++kc) {
      v16bf a = (kc < 2) ? load_a_f32(pd, kc * 32, half) : load_a_f32(ps, kc * 32 - 64, half);
      acc = wmma_bf16(a, load_bp(W0, 4, ct, kc, lane), acc);
    }
    float bb = b0[col];
#pragma unroll
    for (int v = 0; v < 8; ++v)
      smA[(v + 8 * half) * LDP + col] = fmaxf(acc[v] + bb, 0.f);
  }
  __syncthreads();
  const float* xr = smA + cl * LDP;
  for (int ct = 0; ct < 4; ++ct) {
    v8f acc = vzero8();
    for (int kc = 0; kc < 2; ++kc)
      acc = wmma_bf16(load_a_f32(xr, kc * 32, half), load_bp(W1, 2, ct, kc, lane), acc);
    int col = ct * 16 + cl;
    float bb = b1[col];
    if (full) {
#pragma unroll
      for (int v = 0; v < 8; ++v)
        atomicAdd(&Acc[(size_t)di[row0 + v + 8 * half] * 64 + col], acc[v] + bb);
    } else {
#pragma unroll
      for (int v = 0; v < 8; ++v) {
        int r = row0 + v + 8 * half;
        if (r < E) atomicAdd(&Acc[(size_t)di[r] * 64 + col], acc[v] + bb);
      }
    }
  }
}

// 4-layer edge MLP: cat(Xa[ia], Xb[ib]) -> 128:64 relu -> 64:32 relu -> 32:16 relu -> 16:1
__global__ void k_emlp4(const float* Xa, const int* ia, const float* Xb, const int* ib,
                        const bhalf* W0, const float* b0, const bhalf* W1, const float* b1,
                        const bhalf* W2, const float* b2, const float* W3, const float* b3,
                        float* out, int E) {
  __shared__ float smem[4 * 2 * 16 * LDP];
  int wv = threadIdx.x >> 5, lane = threadIdx.x & 31;
  int row0 = (blockIdx.x * 4 + wv) * 16;
  int half = lane >> 4, cl = lane & 15;
  float* smA = smem + (wv * 2) * (16 * LDP);
  float* smB = smA + 16 * LDP;
  int e = row0 + cl; if (e >= E) e = E - 1;
  const float* pa = Xa + (size_t)ia[e] * 64;
  const float* pb = Xb + (size_t)ib[e] * 64;
  for (int ct = 0; ct < 4; ++ct) {                    // L1: 128->64 relu
    v8f acc = vzero8();
    int col = ct * 16 + cl;
    for (int kc = 0; kc < 4; ++kc) {
      v16bf a = (kc < 2) ? load_a_f32(pa, kc * 32, half) : load_a_f32(pb, kc * 32 - 64, half);
      acc = wmma_bf16(a, load_bp(W0, 4, ct, kc, lane), acc);
    }
    float bb = b0[col];
#pragma unroll
    for (int v = 0; v < 8; ++v)
      smA[(v + 8 * half) * LDP + col] = fmaxf(acc[v] + bb, 0.f);
  }
  __syncthreads();
  {                                                   // L2: 64->32 relu
    const float* xr = smA + cl * LDP;
    for (int ct = 0; ct < 2; ++ct) {
      v8f acc = vzero8();
      for (int kc = 0; kc < 2; ++kc)
        acc = wmma_bf16(load_a_f32(xr, kc * 32, half), load_bp(W1, 2, ct, kc, lane), acc);
      int col = ct * 16 + cl;
      float bb = b1[col];
#pragma unroll
      for (int v = 0; v < 8; ++v)
        smB[(v + 8 * half) * LDP + col] = fmaxf(acc[v] + bb, 0.f);
    }
  }
  __syncthreads();
  {                                                   // L3: 32->16 relu
    const float* xr = smB + cl * LDP;
    v8f acc = vzero8();
    acc = wmma_bf16(load_a_f32(xr, 0, half), load_bp(W2, 1, 0, 0, lane), acc);
    int col = cl;
    float bb = b2[col];
#pragma unroll
    for (int v = 0; v < 8; ++v)
      smA[(v + 8 * half) * LDP + col] = fmaxf(acc[v] + bb, 0.f);
  }
  __syncthreads();
  if (lane < 16) {                                    // L4: 16->1 linear
    int ee = row0 + lane;
    if (ee < E) {
      float s = b3[0];
#pragma unroll
      for (int j = 0; j < 16; ++j) s += smA[lane * LDP + j] * W3[j];
      out[ee] = s;
    }
  }
}

// ------------------------------- host side ---------------------------------

extern "C" void kernel_launch(void* const* d_in, const int* in_sizes, int n_in,
                              void* d_out, int out_size, void* d_ws, size_t ws_size,
                              hipStream_t stream) {
  (void)in_sizes; (void)n_in; (void)out_size; (void)ws_size;
  constexpr int N0 = 50000, E0 = 800000, N1 = 5000, E1 = 80000, EF = 200000, EM = 800000;
  const float* x0  = (const float*)d_in[0];
  const float* ea0 = (const float*)d_in[1];
  const float* ea1 = (const float*)d_in[2];
  const float* rv  = (const float*)d_in[3];
  auto P = [&](int leaf) { return (const float*)d_in[4 + leaf]; };
  const int* ei0  = (const int*)d_in[82];
  const int* ei1  = (const int*)d_in[83];
  const int* f2cF = (const int*)d_in[84];
  const int* f2cC = (const int*)d_in[85];
  const int* mrow = (const int*)d_in[86];
  const int* mcol = (const int*)d_in[87];
  const int *s0 = ei0, *d0 = ei0 + E0, *s1 = ei1, *d1 = ei1 + E1;
  float* outp = (float*)d_out;

  // ---- workspace carve-out ----
  char* ws = (char*)d_ws; size_t off = 0;
  auto bump = [&](size_t bytes) -> void* {
    void* p = ws + off; off = (off + bytes + 255) & ~(size_t)255; return p;
  };
  auto AF = [&](size_t n) { return (float*)bump(n * sizeof(float)); };
  auto AB = [&](size_t n) { return (bhalf*)bump(n * sizeof(bhalf)); };

  bhalf* pemW1 = AB(64 * 64);
  bhalf* peW1  = AB(64 * 64);
  bhalf* pnW1  = AB(64 * 64);
  bhalf* pnW2  = AB(64 * 64);
  bhalf *fcW0[2], *fcW1[2], *cfW0[2], *cfW1[2], *tg0[2][3], *tg1[2][3];
  for (int l = 0; l < 2; ++l) {
    fcW0[l] = AB(128 * 64); fcW1[l] = AB(64 * 64);
    cfW0[l] = AB(128 * 64); cfW1[l] = AB(64 * 64);
    for (int k = 0; k < 3; ++k) tg0[l][k] = AB(128 * 64);
    for (int k = 0; k < 3; ++k) tg1[l][k] = AB(128 * 64);
  }
  bhalf* loW  = AB(64 * 64);
  bhalf* locW = AB(64 * 64);
  bhalf* emW0 = AB(128 * 64); bhalf* emW1 = AB(64 * 32); bhalf* emW2 = AB(32 * 16);
  bhalf* erW0 = AB(128 * 64); bhalf* erW1 = AB(64 * 32); bhalf* erW2 = AB(32 * 16);

  float* w0b  = AF((size_t)E0);
  float* w1b  = AF((size_t)E1);
  float* x1b  = AF((size_t)N1);
  float* h0   = AF((size_t)N0 * 64);
  float* h1   = AF((size_t)N1 * 64);
  float* xfc  = AF((size_t)N1 * 64);
  float* xcf  = AF((size_t)N0 * 64);
  float* c0   = AF((size_t)N0 * 128);
  float* c1   = AF((size_t)N1 * 128);
  float* t1f  = AF((size_t)N0 * 128);
  float* t2f  = AF((size_t)N0 * 128);
  float* t1c  = AF((size_t)N1 * 128);
  float* t2c  = AF((size_t)N1 * 128);
  float* xf   = AF((size_t)N0 * 64);
  float* xc   = AF((size_t)N1 * 64);
  float* eRb  = AF((size_t)EF);
  float* csum = AF((size_t)N0);
  float* stats = AF(128);

  auto PACK = [&](const float* s, bhalf* d, int K, int N) {
    int n = K * N;
    k_packB<<<(n + 255) / 256, 256, 0, stream>>>(s, d, K, N);
  };
  auto ZERO = [&](float* p, long long n) { k_zero<<<1024, 256, 0, stream>>>(p, n); };
  auto RELU = [&](float* p, long long n) { k_relu<<<1024, 256, 0, stream>>>(p, n); };
  auto WG = [&](int M) { return ((M + 15) / 16 + 3) / 4; };  // blocks of 4 wave-tiles

  // ---- pre-pack all WMMA weights into B-fragment order ----
  PACK(P(2), pemW1, 64, 64);  PACK(P(8), peW1, 64, 64);
  PACK(P(14), pnW1, 64, 64);  PACK(P(16), pnW2, 64, 64);
  for (int l = 0; l < 2; ++l) {
    int b = 18 + l * 20;
    PACK(P(b + 0), fcW0[l], 128, 64); PACK(P(b + 2), fcW1[l], 64, 64);
    PACK(P(b + 4), cfW0[l], 128, 64); PACK(P(b + 6), cfW1[l], 64, 64);
    PACK(P(b + 8),  tg0[l][0], 128, 64); PACK(P(b + 10), tg0[l][1], 128, 64);
    PACK(P(b + 12), tg0[l][2], 128, 64);
    PACK(P(b + 14), tg1[l][0], 128, 64); PACK(P(b + 16), tg1[l][1], 128, 64);
    PACK(P(b + 18), tg1[l][2], 128, 64);
  }
  PACK(P(58), loW, 64, 64);   PACK(P(60), locW, 64, 64);
  PACK(P(62), emW0, 128, 64); PACK(P(64), emW1, 64, 32); PACK(P(66), emW2, 32, 16);
  PACK(P(70), erW0, 128, 64); PACK(P(72), erW1, 64, 32); PACK(P(74), erW2, 32, 16);

  // ---- edge weights w0 (fused pre_edge_main + pre_edge) and w1 ----
  k_w0_fused<<<WG(E0), 128, 0, stream>>>(ea0,
      P(0), P(1), pemW1, P(3), P(4), P(5),
      P(6), P(7), peW1,  P(9), P(10), P(11), w0b, E0);
  k_pe_fused<<<WG(E1), 128, 0, stream>>>(ea1,
      P(6), P(7), peW1, P(9), P(10), P(11), w1b, E1);

  // ---- coarse node input x1 = R^T x0 ----
  ZERO(x1b, N1);
  k_scat_x1<<<(EF + 255) / 256, 256, 0, stream>>>(x0, rv, f2cF, f2cC, x1b, EF);

  // ---- node embeddings ----
  k_pre_node<<<WG(N0), 128, 0, stream>>>(x0,  P(12), P(13), pnW1, P(15), pnW2, P(17), h0, N0);
  k_pre_node<<<WG(N1), 128, 0, stream>>>(x1b, P(12), P(13), pnW1, P(15), pnW2, P(17), h1, N1);

  // ---- message-passing layers ----
  for (int l = 0; l < 2; ++l) {
    int b = 18 + l * 20;
    ZERO(xfc, (long long)N1 * 64);
    k_plainmp<<<WG(EF), 128, 0, stream>>>(h1, f2cC, h0, f2cF,
        fcW0[l], P(b + 1), fcW1[l], P(b + 3), xfc, EF);
    RELU(xfc, (long long)N1 * 64);
    ZERO(xcf, (long long)N0 * 64);
    k_plainmp<<<WG(EF), 128, 0, stream>>>(h0, f2cF, h1, f2cC,
        cfW0[l], P(b + 5), cfW1[l], P(b + 7), xcf, EF);
    RELU(xcf, (long long)N0 * 64);

    k_concat<<<(int)(((long long)N0 * 128 + 255) / 256), 256, 0, stream>>>(h0, xcf, c0, N0);
    k_concat<<<(int)(((long long)N1 * 128 + 255) / 256), 256, 0, stream>>>(h1, xfc, c1, N1);

    // fine TAGConv (K=2 hops)
    ZERO(t1f, (long long)N0 * 128);
    k_spmv<<<(int)(((long long)E0 * 128 + 255) / 256), 256, 0, stream>>>(c0, s0, d0, w0b, t1f, E0);
    ZERO(t2f, (long long)N0 * 128);
    k_spmv<<<(int)(((long long)E0 * 128 + 255) / 256), 256, 0, stream>>>(t1f, s0, d0, w0b, t2f, E0);
    k_tag_out<<<WG(N0), 128, 0, stream>>>(c0, t1f, t2f,
        tg0[l][0], tg0[l][1], tg0[l][2], P(b + 9), P(b + 11), P(b + 13), h0, N0);
    ZERO(stats, 128);
    k_inorm_stats<<<64, 256, 0, stream>>>(h0, N0, stats);
    k_inorm_apply<<<(int)(((long long)N0 * 64 + 255) / 256), 256, 0, stream>>>(h0, N0, stats);

    // coarse TAGConv
    ZERO(t1c, (long long)N1 * 128);
    k_spmv<<<(int)(((long long)E1 * 128 + 255) / 256), 256, 0, stream>>>(c1, s1, d1, w1b, t1c, E1);
    ZERO(t2c, (long long)N1 * 128);
    k_spmv<<<(int)(((long long)E1 * 128 + 255) / 256), 256, 0, stream>>>(t1c, s1, d1, w1b, t2c, E1);
    k_tag_out<<<WG(N1), 128, 0, stream>>>(c1, t1c, t2c,
        tg1[l][0], tg1[l][1], tg1[l][2], P(b + 15), P(b + 17), P(b + 19), h1, N1);
    ZERO(stats, 128);
    k_inorm_stats<<<64, 256, 0, stream>>>(h1, N1, stats);
    k_inorm_apply<<<(int)(((long long)N1 * 64 + 255) / 256), 256, 0, stream>>>(h1, N1, stats);
  }

  // ---- output projections ----
  k_gemm<<<WG(N0), 128, 0, stream>>>(h0, loW,  P(59), xf, N0, 64, 64, 0);
  k_gemm<<<WG(N1), 128, 0, stream>>>(h1, locW, P(61), xc, N1, 64, 64, 0);

  // ---- out_R: edge_model_R on (xc[f2c_coarse], xf[f2c_fine]) + column norm ----
  k_emlp4<<<WG(EF), 128, 0, stream>>>(xc, f2cC, xf, f2cF,
      erW0, P(71), erW1, P(73), erW2, P(75), P(76), P(77), eRb, EF);
  ZERO(csum, N0);
  k_colsum<<<(EF + 255) / 256, 256, 0, stream>>>(eRb, f2cF, csum, EF);
  k_rnorm<<<(EF + 255) / 256, 256, 0, stream>>>(eRb, f2cF, csum, outp + EM, EF);

  // ---- out: edge_model on (xf[mask_row], xf[mask_col]) ----
  k_emlp4<<<WG(EM), 128, 0, stream>>>(xf, mrow, xf, mcol,
      emW0, P(63), emW1, P(65), emW2, P(67), P(68), P(69), outp, EM);
}
